// soft_margin_triplet_49168785604851
// MI455X (gfx1250) — matrix operand order repaired and checked by reading, not compiled, not run on hardware
//
#include <hip/hip_runtime.h>

// ---------------------------------------------------------------------------
// Histogram-weighted triplet loss, N=8192, D=256, NBINS=64  (MI455X / gfx1250)
// Heavy op: Gram matrix x @ x^T (34.4 GFLOP) -> fused bf16 WMMA + masked
// row max/min of squared distances (sqrt deferred), with async global->LDS
// double-buffered B tiles (ASYNCcnt path). B fragments are preloaded per pass
// so only partial s_wait_dscnt sits between ds_loads and WMMAs.
// ---------------------------------------------------------------------------

#define NROWS 8192
#define DDIM  256
#define NBINS 64
// LDS row pitch in bf16 elements: 256 data + 8 pad = 264 (528 B row stride ->
// 4-bank rotation per row -> conflict-free ds_load_b128 across 16 lanes).
#define PITCH 264

typedef __attribute__((ext_vector_type(16))) __bf16 v16bf;
typedef __attribute__((ext_vector_type(8)))  float  v8f;

union FragBf { v16bf v; uint4 q[2]; };

__device__ __forceinline__ unsigned short f2bf(float f) {
    // round-to-nearest-even f32 -> bf16 (finite gaussian inputs; no NaN path)
    unsigned int u = __float_as_uint(f);
    unsigned int r = u + 0x7FFFu + ((u >> 16) & 1u);
    return (unsigned short)(r >> 16);
}

// Issue an async DMA of a 64-row x 256-col bf16 tile (32 KB) from global xb
// (rows row0..row0+63, unpadded) into a padded LDS buffer. 2048 x 16B chunks,
// 8 per thread. Tracked by ASYNCcnt.
__device__ __forceinline__ void issue_tile_async(
    const unsigned short* __restrict__ xb, int row0,
    unsigned short* ldsbuf, int tid)
{
    unsigned lbase = (unsigned)(unsigned long long)(const void*)ldsbuf;
    #pragma unroll
    for (int i = 0; i < 8; ++i) {
        int ch = tid + i * 256;          // chunk id 0..2047
        int r  = ch >> 5;                // row 0..63
        int c  = ch & 31;                // 16B chunk within row
        unsigned long long g =
            (unsigned long long)(xb + ((size_t)(row0 + r) << 8) + (c << 3));
        unsigned loff = lbase + (unsigned)(r * (PITCH * 2) + (c << 4));
        asm volatile("global_load_async_to_lds_b128 %0, %1, off"
                     :: "v"(loff), "v"(g) : "memory");
    }
}

__device__ __forceinline__ void wait_async0() {
    asm volatile("s_wait_asynccnt 0" ::: "memory");
}

// ---------------------------------------------------------------------------
// k0: per-row squared norm, f32 -> bf16 conversion, accumulator init.
// ---------------------------------------------------------------------------
__global__ __launch_bounds__(256) void k0_prep(
    const float* __restrict__ x,
    unsigned short* __restrict__ xb,
    float* __restrict__ sq,
    unsigned int* __restrict__ posb,
    unsigned int* __restrict__ negb)
{
    const int wave = threadIdx.x >> 5;
    const int lane = threadIdx.x & 31;
    const int row  = blockIdx.x * 8 + wave;

    const float4* xr = (const float4*)(x + (size_t)row * DDIM);
    float4 f0 = xr[lane * 2 + 0];
    float4 f1 = xr[lane * 2 + 1];

    float s = f0.x * f0.x + f0.y * f0.y + f0.z * f0.z + f0.w * f0.w
            + f1.x * f1.x + f1.y * f1.y + f1.z * f1.z + f1.w * f1.w;
    for (int m = 1; m < 32; m <<= 1) s += __shfl_xor(s, m, 32);

    unsigned int p0 = (unsigned int)f2bf(f0.x) | ((unsigned int)f2bf(f0.y) << 16);
    unsigned int p1 = (unsigned int)f2bf(f0.z) | ((unsigned int)f2bf(f0.w) << 16);
    unsigned int p2 = (unsigned int)f2bf(f1.x) | ((unsigned int)f2bf(f1.y) << 16);
    unsigned int p3 = (unsigned int)f2bf(f1.z) | ((unsigned int)f2bf(f1.w) << 16);
    ((uint4*)(xb + (size_t)row * DDIM))[lane] = make_uint4(p0, p1, p2, p3);

    if (lane == 0) {
        sq[row]   = s;
        posb[row] = 0u;           // d^2 clamped >= 1e-12 > 0: ordered-uint atomics
        negb[row] = 0x7F800000u;  // +inf
    }
}

// ---------------------------------------------------------------------------
// k1: fused WMMA Gram tile + masked hardest-pos/neg (squared-dist) reduction.
// Block = 64 rows x 4096 cols (grid.y splits columns). Per 64-col step:
// async-DMA next B tile into LDS while 8 waves (4 M-tiles x 2 N-halves,
// 2 N-subtiles each) run 128 WMMAs on the current tile.
// ---------------------------------------------------------------------------
__global__ __launch_bounds__(256) void k1_gram(
    const unsigned short* __restrict__ xb,
    const float* __restrict__ sq,
    const int* __restrict__ targets,
    unsigned int* __restrict__ posb,
    unsigned int* __restrict__ negb)
{
    __shared__ unsigned short Bs[2][64 * PITCH];  // 2 x 33 KB double buffer

    const int tid      = threadIdx.x;
    const int row_base = blockIdx.x * 64;
    const int cb       = blockIdx.y * (NROWS / 2);

    const int lane   = tid & 31;
    const int wave   = tid >> 5;
    const int mtile  = wave & 3;    // rows [mtile*16, +16)
    const int nthalf = wave >> 2;   // handles N-subtiles {2*nthalf, 2*nthalf+1}
    const int half   = lane >> 4;   // K-half selector per ISA frag layout
    const int l15    = lane & 15;

    // kick off first B tile DMA before doing anything else
    issue_tile_async(xb, cb, Bs[0], tid);

    // A fragments: loaded once from global, kept in 64 VGPRs.
    // A 16x32 bf16 layout: lanes 0-15 K{0..7,16..23}, lanes 16-31 K{8..15,24..31}
    const uint4* arow =
        (const uint4*)(xb + (size_t)(row_base + mtile * 16 + l15) * DDIM);
    FragBf afr[8];
    #pragma unroll
    for (int k8 = 0; k8 < 8; ++k8) {
        afr[k8].q[0] = arow[(k8 * 32 + 8 * half) >> 3];
        afr[k8].q[1] = arow[(k8 * 32 + 16 + 8 * half) >> 3];
    }

    // per-lane row metadata: C/D VGPR r holds row m_local = r + 8*half
    float sq_m[8]; int t_m[8];
    #pragma unroll
    for (int r = 0; r < 8; ++r) {
        int m   = row_base + mtile * 16 + r + 8 * half;
        sq_m[r] = sq[m];
        t_m[r]  = targets[m];
    }
    float pos2[8], neg2[8];
    #pragma unroll
    for (int r = 0; r < 8; ++r) {
        pos2[r] = -__builtin_inff();
        neg2[r] =  __builtin_inff();
    }

    wait_async0();
    __syncthreads();   // first B tile resident

    int cur = 0;
    for (int nb = 0; nb < NROWS / 2; nb += 64) {
        if (nb + 64 < NROWS / 2)
            issue_tile_async(xb, cb + nb + 64, Bs[cur ^ 1], tid);

        #pragma unroll
        for (int p = 0; p < 2; ++p) {
            const int nt  = nthalf * 2 + p;
            const int col = cb + nb + nt * 16 + l15;
            // issue column-metadata loads first: latency hides under WMMAs
            const float sq_n = sq[col];
            const int   t_n  = targets[col];

            // preload all 8 B fragments (one ds_load clause -> partial waits)
            // B 32x16 bf16 layout: lanes 0-15 K{0..7,16..23}, lanes 16-31 K{8..15,24..31}
            const uint4* brow =
                (const uint4*)(&Bs[cur][(size_t)(nt * 16 + l15) * PITCH]);
            FragBf bf[8];
            #pragma unroll
            for (int k8 = 0; k8 < 8; ++k8) {
                bf[k8].q[0] = brow[(k8 * 32 + 8 * half) >> 3];
                bf[k8].q[1] = brow[(k8 * 32 + 16 + 8 * half) >> 3];
            }

            v8f acc = {};
            #pragma unroll
            for (int k8 = 0; k8 < 8; ++k8) {
                acc = __builtin_amdgcn_wmma_f32_16x16x32_bf16(
                    false, afr[k8].v, false, bf[k8].v, (short)0, acc, false, false);
            }

            #pragma unroll
            for (int r = 0; r < 8; ++r) {
                float d2 = sq_m[r] + sq_n - 2.0f * acc[r];  // squared distance
                if (t_m[r] == t_n) pos2[r] = fmaxf(pos2[r], d2);
                else               neg2[r] = fminf(neg2[r], d2);
            }
        }

        wait_async0();     // my next-buffer DMAs landed
        __syncthreads();   // all waves done reading cur; next buffer visible
        cur ^= 1;
    }

    // reduce across the 16 lanes of each half, clamp, merge via ordered atomics
    #pragma unroll
    for (int r = 0; r < 8; ++r) {
        float p = pos2[r], ng = neg2[r];
        for (int m = 1; m <= 8; m <<= 1) {
            p  = fmaxf(p,  __shfl_xor(p,  m, 32));
            ng = fminf(ng, __shfl_xor(ng, m, 32));
        }
        pos2[r] = fmaxf(p,  1e-12f);   // clamp commutes with max/min
        neg2[r] = fmaxf(ng, 1e-12f);
    }
    if (l15 == 0) {
        #pragma unroll
        for (int r = 0; r < 8; ++r) {
            int m = row_base + mtile * 16 + r + 8 * half;
            atomicMax(&posb[m], __float_as_uint(pos2[r]));  // positive: order-safe
            atomicMin(&negb[m], __float_as_uint(neg2[r]));
        }
    }
}

// ---------------------------------------------------------------------------
// k2: histogram + CDF-weighted loss. Single block, deterministic.
// Inputs are squared distances; sqrt applied here (once per row).
// ---------------------------------------------------------------------------
__global__ __launch_bounds__(256) void k2_final(
    const unsigned int* __restrict__ posb,
    const unsigned int* __restrict__ negb,
    float* __restrict__ out)
{
    __shared__ float hvs[NROWS];        // 32 KB
    __shared__ float red[256];
    __shared__ float red2[256];
    __shared__ float hpart[4][NBINS];   // 4-way split bin partials
    __shared__ float hist[NBINS];
    __shared__ float cdfs[NBINS];

    const int tid = threadIdx.x;
    float lmax = -__builtin_inff(), lmin = __builtin_inff();
    for (int i = tid; i < NROWS; i += 256) {
        float hv = sqrtf(__uint_as_float(posb[i])) - sqrtf(__uint_as_float(negb[i]));
        hvs[i] = hv;
        lmax = fmaxf(lmax, hv);
        lmin = fminf(lmin, hv);
    }
    red[tid] = lmax; red2[tid] = lmin;
    __syncthreads();
    for (int s = 128; s > 0; s >>= 1) {
        if (tid < s) {
            red[tid]  = fmaxf(red[tid],  red[tid + s]);
            red2[tid] = fminf(red2[tid], red2[tid + s]);
        }
        __syncthreads();
    }
    const float max_val = fmaxf(red[0], 2.0f);
    const float min_val = fminf(red2[0], -2.0f);
    const float bw      = (max_val - min_val) / (float)(NBINS - 1);
    __syncthreads();

    // deterministic bin-owner histogram, 4-way parallel over rows:
    // thread (g,b) accumulates bin b over rows [g*2048, (g+1)*2048)
    {
        const int g = tid >> 6;       // 0..3
        const int b = tid & (NBINS - 1);
        float acc = 0.0f;
        for (int i = g * (NROWS / 4); i < (g + 1) * (NROWS / 4); ++i) {
            float hv = hvs[i];
            int   lo = (int)floorf((hv - min_val) / bw);
            int   hi = lo + 1;
            hi = hi < 0 ? 0 : (hi > NBINS - 1 ? NBINS - 1 : hi);
            float alpha = 1.0f - (hv - min_val - (float)lo * bw) / bw;
            if (lo == b) acc += alpha;
            if (hi == b) acc += 1.0f - alpha;
        }
        hpart[g][b] = acc;
    }
    __syncthreads();
    if (tid < NBINS)
        hist[tid] = ((hpart[0][tid] + hpart[1][tid]) + hpart[2][tid]) + hpart[3][tid];
    __syncthreads();

    if (tid == 0) {
        float sum = 0.0f;
        for (int b = 0; b < NBINS; ++b) sum += hist[b];
        const float inv1 = 1.0f / (sum + 1e-6f);
        float sum2 = 0.0f;
        for (int b = 0; b < NBINS; ++b) sum2 += hist[b] * inv1;
        float c = 0.0f;
        for (int b = 0; b < NBINS; ++b) {
            c += (hist[b] * inv1) / sum2;
            cdfs[b] = c;
        }
    }
    __syncthreads();

    // loss = mean((pos-neg) * cdf[bin_idx]) == -mean(neg*w) + mean(pos*w)
    float part = 0.0f;
    for (int i = tid; i < NROWS; i += 256) {
        float hv = hvs[i];
        int   bi = (int)floorf((hv - min_val) / bw);
        part += hv * cdfs[bi];
    }
    red[tid] = part;
    __syncthreads();
    for (int s = 128; s > 0; s >>= 1) {
        if (tid < s) red[tid] += red[tid + s];
        __syncthreads();
    }
    if (tid == 0) out[0] = red[0] / (float)NROWS;
}

// ---------------------------------------------------------------------------
extern "C" void kernel_launch(void* const* d_in, const int* in_sizes, int n_in,
                              void* d_out, int out_size, void* d_ws, size_t ws_size,
                              hipStream_t stream) {
    const float* x       = (const float*)d_in[0];
    const int*   targets = (const int*)d_in[1];
    // d_in[2] (histogram buffer) is unused: first-call momentum == 1.0
    float* out = (float*)d_out;

    char* ws = (char*)d_ws;
    unsigned short* xb   = (unsigned short*)ws;                                   // 4 MB bf16 x
    float*          sq   = (float*)(ws + (size_t)NROWS * DDIM * 2);               // 32 KB
    unsigned int*   posb = (unsigned int*)(ws + (size_t)NROWS * DDIM * 2 + NROWS * 4);
    unsigned int*   negb = (unsigned int*)(ws + (size_t)NROWS * DDIM * 2 + NROWS * 8);

    k0_prep<<<NROWS / 8, 256, 0, stream>>>(x, xb, sq, posb, negb);
    k1_gram<<<dim3(NROWS / 64, 2), 256, 0, stream>>>(xb, sq, targets, posb, negb);
    k2_final<<<1, 256, 0, stream>>>(posb, negb, out);
}
